// CrossAtt_27711128994442
// MI455X (gfx1250) — matrix operand order repaired and checked by reading, not compiled, not run on hardware
//
#include <hip/hip_runtime.h>
#include <hip/hip_bf16.h>

// ---------------------------------------------------------------------------
// CDNA5 (gfx1250) cross-attention fusion.
// All matmuls in bf16 via v_wmma_f32_16x16x32_bf16 (f32 accumulation).
// GEMM: 256 threads / 8 waves, block tile 128(M) x 64(N), K-step 32,
// double-buffered LDS; async global->LDS copies (global_load_async_to_lds_b128).
// ---------------------------------------------------------------------------

typedef __bf16 bf16_t;
typedef __attribute__((ext_vector_type(16))) __bf16 v16bf;
typedef __attribute__((ext_vector_type(8)))  __bf16 v8bf;
typedef __attribute__((ext_vector_type(8)))  float   v8f;
typedef __attribute__((ext_vector_type(4)))  int     v4i;

#define LDT 40  // padded LDS row stride (bf16 elements); 80B rows avoid bank conflicts

#if defined(__HIP_DEVICE_COMPILE__) && __has_builtin(__builtin_amdgcn_global_load_async_to_lds_b128)
#define HAS_ASYNC_LDS 1
#define GLB_AS __attribute__((address_space(1)))
#define LDS_AS __attribute__((address_space(3)))
#else
#define HAS_ASYNC_LDS 0
#endif

static constexpr int  Bc  = 8;
static constexpr int  Cc  = 512;
static constexpr int  Nn  = 4096;   // H*W
static constexpr int  OC  = 64;
static constexpr int  C2  = 1024;   // 2*C

#if HAS_ASYNC_LDS
// 16-byte async copy: global -> LDS (tracked by ASYNCcnt)
__device__ __forceinline__ void async_copy16(const bf16_t* gsrc, bf16_t* ldst) {
  const GLB_AS v4i* g1 = (const GLB_AS v4i*)gsrc;   // addrspacecast (pattern OK'd)
  __builtin_amdgcn_global_load_async_to_lds_b128(
      (GLB_AS v4i*)g1,                              // const-drop within AS1
      (LDS_AS v4i*)ldst, 0, 0);
}
#endif

__device__ __forceinline__ void wait_async_then_barrier() {
#if HAS_ASYNC_LDS
#if __has_builtin(__builtin_amdgcn_s_wait_asynccnt)
  __builtin_amdgcn_s_wait_asynccnt(0);
#else
  asm volatile("s_wait_asynccnt 0" ::: "memory");
#endif
#endif
  __syncthreads();
}

// ----------------------------- elementwise helpers -------------------------

__global__ void f32_to_bf16_kernel(const float* __restrict__ src,
                                   bf16_t* __restrict__ dst, int n) {
  int i = blockIdx.x * blockDim.x + threadIdx.x;
  if (i < n) dst[i] = (bf16_t)src[i];
}

// w_cat [64][1024][3][3] (f32) -> tap-major bf16 [9][64][1024]
__global__ void wcat_transform_kernel(const float* __restrict__ w,
                                      bf16_t* __restrict__ wt) {
  int i = blockIdx.x * blockDim.x + threadIdx.x;   // over OC*C2*9
  if (i >= OC * C2 * 9) return;
  int tap = i % 9, tmp = i / 9;
  int ci = tmp % C2, o = tmp / C2;
  wt[((size_t)tap * OC + o) * C2 + ci] = (bf16_t)w[i];
}

// Convert inputs to bf16 and compute per-(b,c) spatial mean. grid=(B*C, 2)
__global__ void __launch_bounds__(256)
convert_pool_kernel(const float* __restrict__ x1, const float* __restrict__ x2,
                    bf16_t* __restrict__ xb1, bf16_t* __restrict__ xb2,
                    float* __restrict__ pooled /* [2][B*C] */) {
  const int which = blockIdx.y;
  const float* src = which ? x2 : x1;
  bf16_t*      dst = which ? xb2 : xb1;
  const size_t base = (size_t)blockIdx.x * Nn;
  float s = 0.f;
  for (int i = threadIdx.x; i < Nn; i += 256) {
    float v = src[base + i];
    dst[base + i] = (bf16_t)v;
    s += v;
  }
  __shared__ float red[256];
  red[threadIdx.x] = s; __syncthreads();
  for (int st = 128; st > 0; st >>= 1) {
    if (threadIdx.x < st) red[threadIdx.x] += red[threadIdx.x + st];
    __syncthreads();
  }
  if (threadIdx.x == 0) pooled[which * (Bc * Cc) + blockIdx.x] = red[0] * (1.0f / Nn);
}

// gate[which][b] = sigmoid(sum_c pooled[which][b][c] * w_lin[c]). grid=(B,2)
__global__ void __launch_bounds__(256)
gate_kernel(const float* __restrict__ pooled, const float* __restrict__ w_lin,
            float* __restrict__ gates /* [2][B] */) {
  const int which = blockIdx.y, b = blockIdx.x;
  float s = 0.f;
  for (int c = threadIdx.x; c < Cc; c += 256)
    s += pooled[which * (Bc * Cc) + b * Cc + c] * w_lin[c];
  __shared__ float red[256];
  red[threadIdx.x] = s; __syncthreads();
  for (int st = 128; st > 0; st >>= 1) {
    if (threadIdx.x < st) red[threadIdx.x] += red[threadIdx.x + st];
    __syncthreads();
  }
  if (threadIdx.x == 0) gates[which * Bc + b] = 1.f / (1.f + __expf(-red[0]));
}

// Row softmax over 512 elements, f32 in -> bf16 out. grid = B*C rows.
__global__ void __launch_bounds__(256)
softmax_kernel(const float* __restrict__ attn, bf16_t* __restrict__ attn_b) {
  const size_t row = (size_t)blockIdx.x * Cc;
  float v0 = attn[row + threadIdx.x];
  float v1 = attn[row + threadIdx.x + 256];
  __shared__ float red[256];
  red[threadIdx.x] = fmaxf(v0, v1); __syncthreads();
  for (int st = 128; st > 0; st >>= 1) {
    if (threadIdx.x < st) red[threadIdx.x] = fmaxf(red[threadIdx.x], red[threadIdx.x + st]);
    __syncthreads();
  }
  float m = red[0]; __syncthreads();
  float e0 = __expf(v0 - m), e1 = __expf(v1 - m);
  red[threadIdx.x] = e0 + e1; __syncthreads();
  for (int st = 128; st > 0; st >>= 1) {
    if (threadIdx.x < st) red[threadIdx.x] += red[threadIdx.x + st];
    __syncthreads();
  }
  float inv = 1.f / red[0];
  attn_b[row + threadIdx.x]       = (bf16_t)(e0 * inv);
  attn_b[row + threadIdx.x + 256] = (bf16_t)(e1 * inv);
}

// ----------------------------- WMMA GEMM -----------------------------------
// D[b] (M x N) = A (M x K) * B[b] (K x N)  [+ bias]  [gate/residual epilogue]
// 256 threads = 8 waves. Block tile 128(M) x 64(N), K step 32, double buffer.
// TA: A stored K-major (stage transposed).  TB: B stored N-major rows (direct).
// EPI: 0 = f32 store, 1 = +bias -> bf16, 2 = gate*acc + resid -> bf16.
template<bool TA, bool TB, int EPI>
__global__ void __launch_bounds__(256)
gemm_wmma_kernel(const bf16_t* __restrict__ A, int lda, long long sA,
                 const bf16_t* __restrict__ Bm, int ldb, long long sB,
                 void* __restrict__ Out, int ldo, long long sO,
                 const float* __restrict__ bias,
                 const float* __restrict__ resid, long long sR,
                 const float* __restrict__ gate,
                 int M, int N, int K) {
  __shared__ bf16_t lA[2][128 * LDT];
  __shared__ bf16_t lB[2][64 * LDT];   // transposed tile: lB[.][n][k]
  const int tid  = threadIdx.x;
  const int lane = tid & 31;
  const int wave = tid >> 5;        // 0..7 : M sub-tile
  const int bz   = blockIdx.z;
  const int n0   = blockIdx.x * 64;
  const int m0   = blockIdx.y * 128;
  const bf16_t* Ab = A  + (size_t)bz * sA;
  const bf16_t* Bb = Bm + (size_t)bz * sB;

  v8f zero = {};
  v8f acc[4] = {zero, zero, zero, zero};

  auto stage = [&](int sb, int kk) {
    // ---- A tile lA[sb][m][k], 128x32 ----
    if (!TA) {
      int m = tid >> 1, ko = (tid & 1) * 16;
      const bf16_t* src = Ab + (size_t)(m0 + m) * lda + kk + ko;
      bf16_t* dst = &lA[sb][m * LDT + ko];
#if HAS_ASYNC_LDS
      async_copy16(src,     dst);
      async_copy16(src + 8, dst + 8);
#else
      *(v8bf*)dst       = *(const v8bf*)(src);
      *(v8bf*)(dst + 8) = *(const v8bf*)(src + 8);
#endif
    } else {
      int k = tid >> 3, mo = (tid & 7) * 16;
      const bf16_t* src = Ab + (size_t)(kk + k) * lda + m0 + mo;
      v8bf t0 = *(const v8bf*)(src);
      v8bf t1 = *(const v8bf*)(src + 8);
      #pragma unroll
      for (int i = 0; i < 8; ++i) {
        lA[sb][(mo + i)     * LDT + k] = t0[i];
        lA[sb][(mo + 8 + i) * LDT + k] = t1[i];
      }
    }
    // ---- B tile transposed lB[sb][n][k], 64x32 ----
    if (TB) {
      int n = tid >> 2, ko8 = (tid & 3) * 8;
      const bf16_t* src = Bb + (size_t)(n0 + n) * ldb + kk + ko8;
      bf16_t* dst = &lB[sb][n * LDT + ko8];
#if HAS_ASYNC_LDS
      async_copy16(src, dst);
#else
      *(v8bf*)dst = *(const v8bf*)(src);
#endif
    } else {
      int k = tid >> 3, no = (tid & 7) * 8;
      const bf16_t* src = Bb + (size_t)(kk + k) * ldb + n0 + no;
      v8bf t0 = *(const v8bf*)(src);
      #pragma unroll
      for (int i = 0; i < 8; ++i) lB[sb][(no + i) * LDT + k] = t0[i];
    }
  };

  stage(0, 0);
  wait_async_then_barrier();

  int buf = 0;
  for (int kk = 0; kk < K; kk += 32) {
    if (kk + 32 < K) stage(buf ^ 1, kk + 32);

    // ---- fragments (per ISA 16-bit A / B VGPR layouts) + 4 WMMAs ----
    const int r = lane & 15, half = lane >> 4;
    v8bf a0 = *(const v8bf*)&lA[buf][(wave * 16 + r) * LDT + half * 8];
    v8bf a1 = *(const v8bf*)&lA[buf][(wave * 16 + r) * LDT + 16 + half * 8];
    v16bf a;
    #pragma unroll
    for (int i = 0; i < 8; ++i) { a[i] = a0[i]; a[8 + i] = a1[i]; }
    #pragma unroll
    for (int j = 0; j < 4; ++j) {
      v8bf b0 = *(const v8bf*)&lB[buf][(j * 16 + r) * LDT + half * 8];
      v8bf b1 = *(const v8bf*)&lB[buf][(j * 16 + r) * LDT + 16 + half * 8];
      v16bf bb;
      #pragma unroll
      for (int i = 0; i < 8; ++i) { bb[i] = b0[i]; bb[8 + i] = b1[i]; }
      acc[j] = __builtin_amdgcn_wmma_f32_16x16x32_bf16(
          false, a, false, bb, (short)0, acc[j], false, false);
    }
    wait_async_then_barrier();
    buf ^= 1;
  }

  // ---- epilogue: C/D layout lane = (g, n), VGPR r -> row g*8+r ----
  const int g = lane >> 4, cn = lane & 15;
  #pragma unroll
  for (int j = 0; j < 4; ++j) {
    int col = n0 + j * 16 + cn;
    #pragma unroll
    for (int rr = 0; rr < 8; ++rr) {
      int row = m0 + wave * 16 + g * 8 + rr;
      float v = acc[j][rr];
      if (EPI == 0) {
        ((float*)Out)[(size_t)bz * sO + (size_t)row * ldo + col] = v;
      } else if (EPI == 1) {
        v += bias[row];
        ((bf16_t*)Out)[(size_t)bz * sO + (size_t)row * ldo + col] = (bf16_t)v;
      } else {
        float res = resid[(size_t)bz * sR + (size_t)row * ldo + col];
        ((bf16_t*)Out)[(size_t)bz * sO + (size_t)row * ldo + col] =
            (bf16_t)(v * gate[bz] + res);
      }
    }
  }
}

// ----------------------------- conv3x3 (WMMA) ------------------------------
// y[b][o][h][w] = sum_{ci,dy,dx} wcat[o][ci][dy][dx]*concat[b][ci][h+dy-1][w+dx-1]
// N-tile = one image row (64 px) with a 1-px halo in LDS; per (dy, ci-chunk)
// the unshifted row is staged once, and the 3 dx taps are 3 WMMA passes at
// shifted LDS row offsets. Weights come tap-major [9][64][1024].
// grid=(64,1,B), 128 threads = 4 waves (M = 64 out channels).
__global__ void __launch_bounds__(128)
conv3x3_wmma_kernel(const bf16_t* __restrict__ WcT, // [9][64][1024]
                    const bf16_t* __restrict__ Xc,  // [B][1024][64][64]
                    float* __restrict__ Y) {        // [B][64][64][64]
  __shared__ bf16_t lA[3 * 64 * LDT];
  __shared__ bf16_t lB[66 * LDT];   // halo rows 0 and 65 stay zero
  const int tid = threadIdx.x, lane = tid & 31, wave = tid >> 5;
  const int h = blockIdx.x, bz = blockIdx.z;
  v8f zero = {};
  v8f acc[4] = {zero, zero, zero, zero};

  for (int i = tid; i < LDT; i += 128) {      // zero halo rows once
    lB[i]            = (bf16_t)0.0f;
    lB[65 * LDT + i] = (bf16_t)0.0f;
  }

  for (int dy = 0; dy < 3; ++dy) {
    const int hh = h + dy - 1;
    const bool rowok = (hh >= 0) && (hh < 64);
    for (int c0 = 0; c0 < C2; c0 += 32) {
      { // A: three dx tiles, contiguous along ci in tap-major layout
        int o = tid >> 1, ko = (tid & 1) * 16;
        #pragma unroll
        for (int dx = 0; dx < 3; ++dx) {
          const bf16_t* src = WcT + ((size_t)(dy * 3 + dx) * OC + o) * C2 + c0 + ko;
          *(v8bf*)&lA[(dx * 64 + o) * LDT + ko]     = *(const v8bf*)(src);
          *(v8bf*)&lA[(dx * 64 + o) * LDT + ko + 8] = *(const v8bf*)(src + 8);
        }
      }
      { // B: lB[1 + w][k] = rowok ? Xc[b][c0+k][hh][w] : 0   (vector loads)
        int k = tid >> 2, wo = (tid & 3) * 16;
        if (rowok) {
          const bf16_t* src = Xc + (((size_t)bz * C2 + c0 + k) * 64 + hh) * 64 + wo;
          v8bf t0 = *(const v8bf*)(src);
          v8bf t1 = *(const v8bf*)(src + 8);
          #pragma unroll
          for (int i = 0; i < 8; ++i) {
            lB[(1 + wo + i)     * LDT + k] = t0[i];
            lB[(1 + wo + 8 + i) * LDT + k] = t1[i];
          }
        } else {
          #pragma unroll
          for (int i = 0; i < 16; ++i) lB[(1 + wo + i) * LDT + k] = (bf16_t)0.0f;
        }
      }
      __syncthreads();

      const int r = lane & 15, half = lane >> 4;
      #pragma unroll
      for (int dx = 0; dx < 3; ++dx) {
        v8bf a0 = *(const v8bf*)&lA[(dx * 64 + wave * 16 + r) * LDT + half * 8];
        v8bf a1 = *(const v8bf*)&lA[(dx * 64 + wave * 16 + r) * LDT + 16 + half * 8];
        v16bf a;
        #pragma unroll
        for (int i = 0; i < 8; ++i) { a[i] = a0[i]; a[8 + i] = a1[i]; }
        #pragma unroll
        for (int j = 0; j < 4; ++j) {
          int rowb = j * 16 + r + dx;   // halo-shifted w index
          v8bf b0 = *(const v8bf*)&lB[rowb * LDT + half * 8];
          v8bf b1 = *(const v8bf*)&lB[rowb * LDT + 16 + half * 8];
          v16bf bb;
          #pragma unroll
          for (int i = 0; i < 8; ++i) { bb[i] = b0[i]; bb[8 + i] = b1[i]; }
          acc[j] = __builtin_amdgcn_wmma_f32_16x16x32_bf16(
              false, a, false, bb, (short)0, acc[j], false, false);
        }
      }
      __syncthreads();
    }
  }
  const int g = lane >> 4, cn = lane & 15;
  #pragma unroll
  for (int j = 0; j < 4; ++j) {
    int w = j * 16 + cn;
    #pragma unroll
    for (int rr = 0; rr < 8; ++rr) {
      int o = wave * 16 + g * 8 + rr;
      Y[(((size_t)bz * OC + o) * 64 + h) * 64 + w] = acc[j][rr];
    }
  }
}

// ----------------------------- batch norm ----------------------------------

__global__ void __launch_bounds__(256)
bn_stats_kernel(const float* __restrict__ Y, float* __restrict__ stats) {
  const int o = blockIdx.x;  // 64 channels
  float s = 0.f, sq = 0.f;
  for (int i = threadIdx.x; i < Bc * Nn; i += 256) {
    int bb = i >> 12, p = i & (Nn - 1);
    float v = Y[(((size_t)bb * OC + o) << 12) + p];
    s += v; sq += v * v;
  }
  __shared__ float r1[256], r2[256];
  r1[threadIdx.x] = s; r2[threadIdx.x] = sq; __syncthreads();
  for (int st = 128; st > 0; st >>= 1) {
    if (threadIdx.x < st) { r1[threadIdx.x] += r1[threadIdx.x + st];
                            r2[threadIdx.x] += r2[threadIdx.x + st]; }
    __syncthreads();
  }
  if (threadIdx.x == 0) {
    float mean = r1[0] * (1.f / (Bc * Nn));
    float var  = r2[0] * (1.f / (Bc * Nn)) - mean * mean;
    stats[o * 2] = mean; stats[o * 2 + 1] = var;
  }
}

__global__ void bn_apply_kernel(const float* __restrict__ Y,
                                const float* __restrict__ stats,
                                const float* __restrict__ gamma,
                                const float* __restrict__ beta,
                                float* __restrict__ Out) {
  size_t i = (size_t)blockIdx.x * blockDim.x + threadIdx.x;  // B*64*4096
  int o = (int)((i >> 12) & (OC - 1));
  float mean = stats[o * 2], var = stats[o * 2 + 1];
  float v = (Y[i] - mean) * rsqrtf(var + 1e-5f) * gamma[o] + beta[o];
  Out[i] = v > 0.f ? v : 0.f;
}

// ----------------------------- launch --------------------------------------

extern "C" void kernel_launch(void* const* d_in, const int* in_sizes, int n_in,
                              void* d_out, int out_size, void* d_ws, size_t ws_size,
                              hipStream_t stream) {
  const float* input1 = (const float*)d_in[0];
  const float* input2 = (const float*)d_in[1];
  const float* wq[2]  = {(const float*)d_in[2],  (const float*)d_in[8]};   // wq1, wq2
  const float* bq[2]  = {(const float*)d_in[3],  (const float*)d_in[9]};
  const float* wk[2]  = {(const float*)d_in[4],  (const float*)d_in[10]};  // wk1, wk2
  const float* bk[2]  = {(const float*)d_in[5],  (const float*)d_in[11]};
  const float* wv[2]  = {(const float*)d_in[6],  (const float*)d_in[12]};  // wv1, wv2
  const float* bv[2]  = {(const float*)d_in[7],  (const float*)d_in[13]};
  const float* w_lin  = (const float*)d_in[14];
  const float* w_cat  = (const float*)d_in[15];
  const float* gamma  = (const float*)d_in[16];
  const float* beta   = (const float*)d_in[17];

  const long long CN = (long long)Cc * Nn;   // 2,097,152

  // workspace carve-up (256B aligned)
  char* ws = (char*)d_ws;
  size_t off = 0;
  auto take = [&](size_t bytes) -> void* {
    void* p = ws + off; off = (off + bytes + 255) & ~(size_t)255; return p;
  };
  bf16_t* xb1    = (bf16_t*)take((size_t)Bc * CN * 2);
  bf16_t* xb2    = (bf16_t*)take((size_t)Bc * CN * 2);
  bf16_t* wqb[2] = {(bf16_t*)take(Cc * Cc * 2), (bf16_t*)take(Cc * Cc * 2)};
  bf16_t* wkb[2] = {(bf16_t*)take(Cc * Cc * 2), (bf16_t*)take(Cc * Cc * 2)};
  bf16_t* wvb[2] = {(bf16_t*)take(Cc * Cc * 2), (bf16_t*)take(Cc * Cc * 2)};
  bf16_t* wcatT  = (bf16_t*)take((size_t)9 * OC * C2 * 2);
  bf16_t* qb     = (bf16_t*)take((size_t)Bc * CN * 2);
  bf16_t* kb     = (bf16_t*)take((size_t)Bc * CN * 2);
  bf16_t* vb     = (bf16_t*)take((size_t)Bc * CN * 2);
  float*  attnf  = (float*)take((size_t)Bc * Cc * Cc * 4);
  bf16_t* attnbb = (bf16_t*)take((size_t)Bc * Cc * Cc * 2);
  bf16_t* concat = (bf16_t*)take((size_t)Bc * C2 * Nn * 2);
  float*  ybuf   = (float*)take((size_t)Bc * OC * Nn * 4);
  float*  pooled = (float*)take(2 * Bc * Cc * 4);
  float*  gates  = (float*)take(2 * Bc * 4);
  float*  stats  = (float*)take(OC * 2 * 4);
  (void)ws_size; (void)n_in; (void)in_sizes; (void)out_size;

  // 1) fp32 -> bf16 inputs + spatial means
  convert_pool_kernel<<<dim3(Bc * Cc, 2), 256, 0, stream>>>(
      input1, input2, xb1, xb2, pooled);
  // 2) sigmoid gates
  gate_kernel<<<dim3(Bc, 2), 256, 0, stream>>>(pooled, w_lin, gates);
  // 3) weights -> bf16 (+ tap-major transform for w_cat)
  const int nw = Cc * Cc;
  for (int i = 0; i < 2; ++i) {
    f32_to_bf16_kernel<<<(nw + 255) / 256, 256, 0, stream>>>(wq[i], wqb[i], nw);
    f32_to_bf16_kernel<<<(nw + 255) / 256, 256, 0, stream>>>(wk[i], wkb[i], nw);
    f32_to_bf16_kernel<<<(nw + 255) / 256, 256, 0, stream>>>(wv[i], wvb[i], nw);
  }
  const int nwc = OC * C2 * 9;
  wcat_transform_kernel<<<(nwc + 255) / 256, 256, 0, stream>>>(w_cat, wcatT);

  // 4) two attention branches (sequential; scratch reused)
  for (int br = 0; br < 2; ++br) {
    // branch 0: q = wq2(x2), k = wk1(x1), v = wv1(x1), resid/gate x1
    // branch 1: q = wq1(x1), k = wk2(x2), v = wv2(x2), resid/gate x2
    bf16_t* qsrc = (br == 0) ? xb2 : xb1;
    bf16_t* kvs  = (br == 0) ? xb1 : xb2;
    bf16_t* wqm  = (br == 0) ? wqb[1] : wqb[0];
    const float* bqm = (br == 0) ? bq[1] : bq[0];
    bf16_t* wkm  = (br == 0) ? wkb[0] : wkb[1];
    const float* bkm = (br == 0) ? bk[0] : bk[1];
    bf16_t* wvm  = (br == 0) ? wvb[0] : wvb[1];
    const float* bvm = (br == 0) ? bv[0] : bv[1];
    const float* resid = (br == 0) ? input1 : input2;

    dim3 gQKV(Nn / 64, Cc / 128, Bc);
    // q/k/v = W * X + b   (M=512, N=4096, K=512)
    gemm_wmma_kernel<false, false, 1><<<gQKV, 256, 0, stream>>>(
        wqm, Cc, 0, qsrc, Nn, CN, qb, Nn, CN, bqm, nullptr, 0, nullptr, Cc, Nn, Cc);
    gemm_wmma_kernel<false, false, 1><<<gQKV, 256, 0, stream>>>(
        wkm, Cc, 0, kvs, Nn, CN, kb, Nn, CN, bkm, nullptr, 0, nullptr, Cc, Nn, Cc);
    gemm_wmma_kernel<false, false, 1><<<gQKV, 256, 0, stream>>>(
        wvm, Cc, 0, kvs, Nn, CN, vb, Nn, CN, bvm, nullptr, 0, nullptr, Cc, Nn, Cc);

    // attn[kc][c] = sum_n k[kc][n] q[c][n]  (M=512, N=512, K=4096)
    gemm_wmma_kernel<false, true, 0><<<dim3(Cc / 64, Cc / 128, Bc), 256, 0, stream>>>(
        kb, Nn, CN, qb, Nn, CN, attnf, Cc, (long long)Cc * Cc,
        nullptr, nullptr, 0, nullptr, Cc, Cc, Nn);

    // softmax over last dim -> bf16
    softmax_kernel<<<Bc * Cc, 256, 0, stream>>>(attnf, attnbb);

    // out[c][n] = sum_k attn[k][c] v[k][n]; then gate*out + resid -> concat slice
    gemm_wmma_kernel<true, false, 2><<<gQKV, 256, 0, stream>>>(
        attnbb, Cc, (long long)Cc * Cc, vb, Nn, CN,
        concat + (size_t)br * Cc * Nn, Nn, (long long)C2 * Nn,
        nullptr, resid, CN, gates + br * Bc, Cc, Nn, Cc);
  }

  // 5) conv3x3 over concat -> ybuf
  conv3x3_wmma_kernel<<<dim3(64, 1, Bc), 128, 0, stream>>>(wcatT, concat, ybuf);

  // 6) batchnorm (batch stats) + relu -> d_out
  bn_stats_kernel<<<OC, 256, 0, stream>>>(ybuf, stats);
  bn_apply_kernel<<<(Bc * OC * Nn) / 256, 256, 0, stream>>>(
      ybuf, stats, gamma, beta, (float*)d_out);
}
